// MPNN_35897336660050
// MI455X (gfx1250) — compile-verified, matrix operand order checked by reading, not run on hardware
//
#include <hip/hip_runtime.h>

// ---------------------------------------------------------------------------
// MPNN forward for MI455X (gfx1250, wave32, WMMA bf16 16x16x32)
// ---------------------------------------------------------------------------

#define ATOM_FDIM 133
#define BOND_FDIM 14
#define HIDDEN    128
#define MAX_NB    6

typedef __bf16 v16bf __attribute__((ext_vector_type(16)));
typedef float  v8f   __attribute__((ext_vector_type(8)));

struct __attribute__((aligned(16))) U128 { unsigned int x, y, z, w; };
struct __attribute__((aligned(8)))  U64  { unsigned int x, y; };

union Frag16 { U128 u4[2]; v16bf v; };

__device__ __forceinline__ unsigned short f2bf(float f) {
    unsigned int u = __float_as_uint(f);
    unsigned int r = (u + 0x7FFFu + ((u >> 16) & 1u)) >> 16;
    return (unsigned short)r;
}
__device__ __forceinline__ float bf2f(unsigned short h) {
    return __uint_as_float(((unsigned int)h) << 16);
}

// ---------------------------------------------------------------------------
// Pack a row-major fp32 weight matrix W[128][Kin] into bf16 WMMA B-fragments:
// out[ntile][kstep][lane][i], lane holds column ntile*16+(lane&15),
// halfs i map K per the 16-bit operand layout (K-half = lane>>4).
// ---------------------------------------------------------------------------
__global__ void weight_frag_kernel(const float* __restrict__ W, int Kin, int ksteps,
                                   unsigned short* __restrict__ out, int total) {
    int tid = blockIdx.x * blockDim.x + threadIdx.x;
    if (tid >= total) return;
    int i     = tid & 15;
    int lane  = (tid >> 4) & 31;
    int blk   = tid >> 9;            // ntile*ksteps + ks
    int ks    = blk % ksteps;
    int ntile = blk / ksteps;
    int o     = ntile * 16 + (lane & 15);
    int khalf = lane >> 4;
    int g = i >> 3, j = i & 7;
    int k = ks * 32 + g * 16 + khalf * 8 + j;
    float v = (k < Kin) ? W[(size_t)o * Kin + k] : 0.0f;
    out[tid] = f2bf(v);
}

// ---------------------------------------------------------------------------
// nei_f_bonds_sum: sum 6 gathered bond rows -> bf16 [N][16] (padded)
// ---------------------------------------------------------------------------
__global__ void bond_sum_kernel(const float* __restrict__ f_bonds,
                                const int* __restrict__ a2b,
                                unsigned short* __restrict__ bond16,
                                int n_atoms) {
    int tid = blockIdx.x * blockDim.x + threadIdx.x;
    int a = tid >> 4, c = tid & 15;
    if (a >= n_atoms) return;
    float acc = 0.f;
    if (c < BOND_FDIM) {
        #pragma unroll
        for (int nb = 0; nb < MAX_NB; ++nb) {
            int b = a2b[a * MAX_NB + nb];
            acc += f_bonds[(size_t)b * BOND_FDIM + c];
        }
    }
    bond16[(size_t)a * 16 + c] = f2bf(acc);
}

// ---------------------------------------------------------------------------
// K1: x = f_atoms @ Wi^T + b (stored bf16; ReLU is applied lazily by the
// depth-1 gather, so no separate message buffer is written here).
// KPAD = 160 (5 k-steps). 256 threads = 8 waves; wave w -> N-tile w.
// ---------------------------------------------------------------------------
__global__ __launch_bounds__(256)
void input_gemm_kernel(const float* __restrict__ f_atoms,
                       const unsigned short* __restrict__ wiFrag,
                       const float* __restrict__ Wi_b,
                       unsigned short* __restrict__ xbf,
                       int n_atoms) {
    __shared__ unsigned short As[16 * 160];
    const int abase = blockIdx.x * 16;
    const int tid = threadIdx.x;
    {   // build A tile: 16 atoms x 160 (pad 133 -> 160) in bf16
        int a = tid >> 4, p = tid & 15;
        int row = (abase + a < n_atoms) ? abase + a : n_atoms - 1;
        const float* src = f_atoms + (size_t)row * ATOM_FDIM;
        #pragma unroll
        for (int q = 0; q < 10; ++q) {
            int k = p * 10 + q;
            As[a * 160 + k] = f2bf((k < ATOM_FDIM) ? src[k] : 0.f);
        }
    }
    __syncthreads();

    const int wave = tid >> 5, lane = tid & 31;
    const int col = lane & 15, khalf = lane >> 4;
    v8f acc = {};
    const unsigned short* bbase = wiFrag + (size_t)(wave * 5) * 512 + lane * 16;
    #pragma unroll
    for (int ks = 0; ks < 5; ++ks) {
        Frag16 fa, fb;
        const unsigned short* ap = &As[(lane & 15) * 160 + ks * 32 + khalf * 8];
        fa.u4[0] = *(const U128*)(ap);
        fa.u4[1] = *(const U128*)(ap + 16);
        const unsigned short* bp = bbase + ks * 512;
        fb.u4[0] = *(const U128*)(bp);
        fb.u4[1] = *(const U128*)(bp + 8);
        acc = __builtin_amdgcn_wmma_f32_16x16x32_bf16(false, fa.v, false, fb.v,
                                                      (short)0, acc, false, false);
    }
    const int hcol = wave * 16 + col;
    const float bias = Wi_b[hcol];
    if (abase + 16 <= n_atoms) {        // uniform full-tile fast path
        #pragma unroll
        for (int r = 0; r < 8; ++r) {
            int row = abase + r + khalf * 8;
            xbf[(size_t)row * HIDDEN + hcol] = f2bf(acc[r] + bias);
        }
    } else {
        #pragma unroll
        for (int r = 0; r < 8; ++r) {
            int row = abase + r + khalf * 8;
            if (row < n_atoms)
                xbf[(size_t)row * HIDDEN + hcol] = f2bf(acc[r] + bias);
        }
    }
}

// ---------------------------------------------------------------------------
// K3: msgOut = relu(x + [sum relu?(msgIn[a2a]), bond_sum] @ Wh^T + b).
// KPAD = 160. relu_gather=1 on depth 1 (msgIn == x, message0 = relu(x)).
// ---------------------------------------------------------------------------
__global__ __launch_bounds__(256)
void mp_gemm_kernel(const unsigned short* __restrict__ msgIn,
                    const unsigned short* __restrict__ bond16,
                    const unsigned short* __restrict__ xbf,
                    const unsigned short* __restrict__ whFrag,
                    const float* __restrict__ Wh_b,
                    const int* __restrict__ a2a,
                    unsigned short* __restrict__ msgOut,
                    int n_atoms, int relu_gather) {
    __shared__ unsigned short As[16 * 160];
    const int abase = blockIdx.x * 16;
    const int tid = threadIdx.x;
    {   // gather-sum 6 neighbor messages (8 channels per thread) + bond tail
        int a = tid >> 4, p = tid & 15;
        int row = (abase + a < n_atoms) ? abase + a : n_atoms - 1;
        float accs[8] = {0, 0, 0, 0, 0, 0, 0, 0};
        if (relu_gather) {
            for (int nb = 0; nb < MAX_NB; ++nb) {
                int idx = a2a[row * MAX_NB + nb];
                U128 v = *(const U128*)&msgIn[(size_t)idx * HIDDEN + p * 8];
                const unsigned short* hp = (const unsigned short*)&v;
                #pragma unroll
                for (int j = 0; j < 8; ++j) accs[j] += fmaxf(bf2f(hp[j]), 0.f);
            }
        } else {
            for (int nb = 0; nb < MAX_NB; ++nb) {
                int idx = a2a[row * MAX_NB + nb];
                U128 v = *(const U128*)&msgIn[(size_t)idx * HIDDEN + p * 8];
                const unsigned short* hp = (const unsigned short*)&v;
                #pragma unroll
                for (int j = 0; j < 8; ++j) accs[j] += bf2f(hp[j]);
            }
        }
        #pragma unroll
        for (int j = 0; j < 8; ++j) As[a * 160 + p * 8 + j] = f2bf(accs[j]);
        if (p == 0) {
            const U128* bsrc = (const U128*)&bond16[(size_t)row * 16];
            *(U128*)&As[a * 160 + 128] = bsrc[0];
            *(U128*)&As[a * 160 + 136] = bsrc[1];
            for (int k = 144; k < 160; ++k) As[a * 160 + k] = 0;
        }
    }
    __syncthreads();

    const int wave = tid >> 5, lane = tid & 31;
    const int col = lane & 15, khalf = lane >> 4;
    v8f acc = {};
    const unsigned short* bbase = whFrag + (size_t)(wave * 5) * 512 + lane * 16;
    #pragma unroll
    for (int ks = 0; ks < 5; ++ks) {
        Frag16 fa, fb;
        const unsigned short* ap = &As[(lane & 15) * 160 + ks * 32 + khalf * 8];
        fa.u4[0] = *(const U128*)(ap);
        fa.u4[1] = *(const U128*)(ap + 16);
        const unsigned short* bp = bbase + ks * 512;
        fb.u4[0] = *(const U128*)(bp);
        fb.u4[1] = *(const U128*)(bp + 8);
        acc = __builtin_amdgcn_wmma_f32_16x16x32_bf16(false, fa.v, false, fb.v,
                                                      (short)0, acc, false, false);
    }
    const int hcol = wave * 16 + col;
    const float bias = Wh_b[hcol];
    if (abase + 16 <= n_atoms) {        // uniform full-tile fast path
        #pragma unroll
        for (int r = 0; r < 8; ++r) {
            int row = abase + r + khalf * 8;
            float xv = bf2f(xbf[(size_t)row * HIDDEN + hcol]);
            msgOut[(size_t)row * HIDDEN + hcol] = f2bf(fmaxf(xv + acc[r] + bias, 0.f));
        }
    } else {
        #pragma unroll
        for (int r = 0; r < 8; ++r) {
            int row = abase + r + khalf * 8;
            if (row < n_atoms) {
                float xv = bf2f(xbf[(size_t)row * HIDDEN + hcol]);
                msgOut[(size_t)row * HIDDEN + hcol] = f2bf(fmaxf(xv + acc[r] + bias, 0.f));
            }
        }
    }
}

// ---------------------------------------------------------------------------
// K4: hidden = relu([f_atoms, sum msg[a2a]] @ Wo^T + b). KPAD = 288 (9 steps).
// ---------------------------------------------------------------------------
__global__ __launch_bounds__(256)
void output_gemm_kernel(const float* __restrict__ f_atoms,
                        const unsigned short* __restrict__ msgIn,
                        const unsigned short* __restrict__ woFrag,
                        const float* __restrict__ Wo_b,
                        const int* __restrict__ a2a,
                        unsigned short* __restrict__ hidden,
                        int n_atoms) {
    __shared__ unsigned short As[16 * 288];
    const int abase = blockIdx.x * 16;
    const int tid = threadIdx.x;
    {
        int a = tid >> 4, p = tid & 15;
        int row = (abase + a < n_atoms) ? abase + a : n_atoms - 1;
        // atom features: k in [0,133)
        const float* src = f_atoms + (size_t)row * ATOM_FDIM;
        #pragma unroll
        for (int q = 0; q < 9; ++q) {
            int k = p * 9 + q;                       // 0..143
            if (k < ATOM_FDIM) As[a * 288 + k] = f2bf(src[k]);
        }
        // gathered message sum: k in [133,261)
        float accs[8] = {0, 0, 0, 0, 0, 0, 0, 0};
        for (int nb = 0; nb < MAX_NB; ++nb) {
            int idx = a2a[row * MAX_NB + nb];
            U128 v = *(const U128*)&msgIn[(size_t)idx * HIDDEN + p * 8];
            const unsigned short* hp = (const unsigned short*)&v;
            #pragma unroll
            for (int j = 0; j < 8; ++j) accs[j] += bf2f(hp[j]);
        }
        #pragma unroll
        for (int j = 0; j < 8; ++j) As[a * 288 + 133 + p * 8 + j] = f2bf(accs[j]);
        // zero tail [261,288)
        if (p == 0)
            for (int k = 261; k < 288; ++k) As[a * 288 + k] = 0;
    }
    __syncthreads();

    const int wave = tid >> 5, lane = tid & 31;
    const int col = lane & 15, khalf = lane >> 4;
    v8f acc = {};
    const unsigned short* bbase = woFrag + (size_t)(wave * 9) * 512 + lane * 16;
    #pragma unroll
    for (int ks = 0; ks < 9; ++ks) {
        Frag16 fa, fb;
        const unsigned short* ap = &As[(lane & 15) * 288 + ks * 32 + khalf * 8];
        fa.u4[0] = *(const U128*)(ap);
        fa.u4[1] = *(const U128*)(ap + 16);
        const unsigned short* bp = bbase + ks * 512;
        fb.u4[0] = *(const U128*)(bp);
        fb.u4[1] = *(const U128*)(bp + 8);
        acc = __builtin_amdgcn_wmma_f32_16x16x32_bf16(false, fa.v, false, fb.v,
                                                      (short)0, acc, false, false);
    }
    const int hcol = wave * 16 + col;
    const float bias = Wo_b[hcol];
    if (abase + 16 <= n_atoms) {        // uniform full-tile fast path
        #pragma unroll
        for (int r = 0; r < 8; ++r) {
            int row = abase + r + khalf * 8;
            hidden[(size_t)row * HIDDEN + hcol] = f2bf(fmaxf(acc[r] + bias, 0.f));
        }
    } else {
        #pragma unroll
        for (int r = 0; r < 8; ++r) {
            int row = abase + r + khalf * 8;
            if (row < n_atoms)
                hidden[(size_t)row * HIDDEN + hcol] = f2bf(fmaxf(acc[r] + bias, 0.f));
        }
    }
}

// ---------------------------------------------------------------------------
// K5: per-mol max over atoms_per_mol atoms, then dot with ffn_w + ffn_b.
// One wave32 per molecule; lane handles 4 channels.
// ---------------------------------------------------------------------------
__global__ __launch_bounds__(256)
void readout_kernel(const unsigned short* __restrict__ hidden,
                    const float* __restrict__ ffn_w,
                    const float* __restrict__ ffn_b,
                    const int* __restrict__ apm_ptr,
                    float* __restrict__ out,
                    int n_mols) {
    const int wave = threadIdx.x >> 5;
    const int lane = threadIdx.x & 31;
    const int mol = blockIdx.x * 8 + wave;
    if (mol >= n_mols) return;
    const int apm = *apm_ptr;
    float mx0 = -3.402823e38f, mx1 = mx0, mx2 = mx0, mx3 = mx0;
    const int abase = mol * apm;
    for (int t = 0; t < apm; ++t) {
        U64 v = *(const U64*)&hidden[(size_t)(abase + t) * HIDDEN + lane * 4];
        const unsigned short* s = (const unsigned short*)&v;
        mx0 = fmaxf(mx0, bf2f(s[0]));
        mx1 = fmaxf(mx1, bf2f(s[1]));
        mx2 = fmaxf(mx2, bf2f(s[2]));
        mx3 = fmaxf(mx3, bf2f(s[3]));
    }
    float sum = mx0 * ffn_w[lane * 4 + 0] + mx1 * ffn_w[lane * 4 + 1] +
                mx2 * ffn_w[lane * 4 + 2] + mx3 * ffn_w[lane * 4 + 3];
    #pragma unroll
    for (int off = 16; off > 0; off >>= 1) sum += __shfl_xor(sum, off, 32);
    if (lane == 0) out[mol] = sum + ffn_b[0];
}

// ---------------------------------------------------------------------------
extern "C" void kernel_launch(void* const* d_in, const int* in_sizes, int n_in,
                              void* d_out, int out_size, void* d_ws, size_t ws_size,
                              hipStream_t stream) {
    const float* f_atoms = (const float*)d_in[0];
    const float* f_bonds = (const float*)d_in[1];
    const float* Wi_w    = (const float*)d_in[2];
    const float* Wi_b    = (const float*)d_in[3];
    const float* Wh_w    = (const float*)d_in[4];
    const float* Wh_b    = (const float*)d_in[5];
    const float* Wo_w    = (const float*)d_in[6];
    const float* Wo_b    = (const float*)d_in[7];
    const float* ffn_w   = (const float*)d_in[8];
    const float* ffn_b   = (const float*)d_in[9];
    const int*   a2a     = (const int*)d_in[10];
    const int*   a2b     = (const int*)d_in[11];
    const int*   apm     = (const int*)d_in[12];

    const int n_atoms = in_sizes[0] / ATOM_FDIM;
    const int n_mols  = out_size;

    // Workspace carve (bf16 intermediates). 'hidden' reuses the x buffer:
    // x is last read in the second Wh depth, before output_gemm writes hidden.
    char* ws = (char*)d_ws;
    size_t off = 0;
    auto carve = [&](size_t bytes) {
        void* p = ws + off;
        off = (off + bytes + 255) & ~(size_t)255;
        return p;
    };
    unsigned short* xbf    = (unsigned short*)carve((size_t)n_atoms * HIDDEN * 2);
    unsigned short* msgA   = (unsigned short*)carve((size_t)n_atoms * HIDDEN * 2);
    unsigned short* msgB   = (unsigned short*)carve((size_t)n_atoms * HIDDEN * 2);
    unsigned short* bond16 = (unsigned short*)carve((size_t)n_atoms * 16 * 2);
    unsigned short* wiFrag = (unsigned short*)carve((size_t)8 * 5 * 512 * 2);
    unsigned short* whFrag = (unsigned short*)carve((size_t)8 * 5 * 512 * 2);
    unsigned short* woFrag = (unsigned short*)carve((size_t)8 * 9 * 512 * 2);
    unsigned short* hidden = xbf;

    // Weight fragment packing (tiny, L2-resident afterwards)
    {
        int t1 = 8 * 5 * 512, t2 = 8 * 9 * 512;
        weight_frag_kernel<<<(t1 + 255) / 256, 256, 0, stream>>>(Wi_w, ATOM_FDIM, 5, wiFrag, t1);
        weight_frag_kernel<<<(t1 + 255) / 256, 256, 0, stream>>>(Wh_w, HIDDEN + BOND_FDIM, 5, whFrag, t1);
        weight_frag_kernel<<<(t2 + 255) / 256, 256, 0, stream>>>(Wo_w, ATOM_FDIM + HIDDEN, 9, woFrag, t2);
    }

    // Bond-feature neighbor sums (constant across depths)
    bond_sum_kernel<<<((size_t)n_atoms * 16 + 255) / 256, 256, 0, stream>>>(
        f_bonds, a2b, bond16, n_atoms);

    const int tiles = (n_atoms + 15) / 16;

    // x = Wi GEMM + bias (message0 = relu(x) realized lazily in depth-1 gather)
    input_gemm_kernel<<<tiles, 256, 0, stream>>>(f_atoms, wiFrag, Wi_b, xbf, n_atoms);

    // two message-passing depths (double-buffered; depth 1 gathers relu(x))
    mp_gemm_kernel<<<tiles, 256, 0, stream>>>(xbf, bond16, xbf, whFrag, Wh_b, a2a, msgB,
                                              n_atoms, /*relu_gather=*/1);
    mp_gemm_kernel<<<tiles, 256, 0, stream>>>(msgB, bond16, xbf, whFrag, Wh_b, a2a, msgA,
                                              n_atoms, /*relu_gather=*/0);

    // atom_hiddens = relu([f_atoms, gather-sum(msg)] @ Wo^T + b)
    output_gemm_kernel<<<tiles, 256, 0, stream>>>(f_atoms, msgA, woFrag, Wo_b, a2a, hidden, n_atoms);

    // per-molecule max readout + FFN head
    readout_kernel<<<(n_mols + 7) / 8, 256, 0, stream>>>(hidden, ffn_w, ffn_b, apm,
                                                         (float*)d_out, n_mols);
}
